// TrXLAgent_25237227831359
// MI455X (gfx1250) — matrix operand order repaired and checked by reading, not compiled, not run on hardware
//
#include <hip/hip_runtime.h>

// ---------------- problem constants ----------------
constexpr int B_  = 32;
constexpr int K_  = 512;
constexpr int L_  = 6;
constexpr int D_  = 2048;
constexpr int H_  = 16;
constexpr int HD_ = 128;
constexpr int OBS_ = 128;
constexpr int A_  = 10;
constexpr int MAX_STEPS_ = 512;

// ---------------- WMMA vector types ----------------
typedef __attribute__((ext_vector_type(16))) __bf16 v16bf;
typedef __attribute__((ext_vector_type(8)))  float  v8f;

union FragAB { v16bf v; uint4 q[2]; };

// ---------------- bf16 helpers ----------------
__device__ __forceinline__ unsigned short f2bf(float f) {
    unsigned int u = __float_as_uint(f);
    unsigned int r = u + 0x7FFFu + ((u >> 16) & 1u);   // round-to-nearest-even
    return (unsigned short)(r >> 16);
}
__device__ __forceinline__ float bf2f(unsigned int s) {
    return __uint_as_float((s & 0xFFFFu) << 16);
}

// ---------------- fp32 -> bf16 convert ----------------
__global__ void cvt_bf16_kernel(const float* __restrict__ in,
                                unsigned short* __restrict__ out, size_t n) {
    size_t i = (size_t)blockIdx.x * blockDim.x + threadIdx.x;
    if (i < n) out[i] = f2bf(in[i]);
}

// ---------------- encoder: h = x @ enc_w.T + enc_b ----------------
__global__ void __launch_bounds__(256)
encoder_kernel(const float* __restrict__ x, const float* __restrict__ ew,
               const float* __restrict__ eb, float* __restrict__ h) {
    int idx = blockIdx.x * 256 + threadIdx.x;          // B_*D_ outputs
    if (idx >= B_ * D_) return;
    int b = idx / D_, d = idx % D_;
    const float* xr = x + (size_t)b * OBS_;
    const float* wr = ew + (size_t)d * OBS_;
    float acc = eb[d];
    #pragma unroll 4
    for (int j = 0; j < OBS_; ++j) acc += xr[j] * wr[j];
    h[idx] = acc;
}

// ---------------- generic bf16 WMMA GEMM:  C[M,N] = A[M,Kd] * W[N,Kd]^T ----
// epilogue: +bias, optional relu, optional +residual; writes f32 and/or bf16.
__global__ void __launch_bounds__(256)
gemm_bf16_wmma(const unsigned short* __restrict__ A, int lda,
               const unsigned short* __restrict__ W, int ldw,
               float* __restrict__ Cf,
               const float* __restrict__ bias,
               const float* __restrict__ resid,
               unsigned short* __restrict__ Cbf,
               int M, int N, int Kd, int relu) {
    int ntN    = N >> 4;
    int ntiles = (M >> 4) * ntN;
    int tile   = blockIdx.x * 8 + (threadIdx.x >> 5);  // one 16x16 tile per wave
    if (tile >= ntiles) return;
    int m0 = (tile / ntN) << 4;
    int n0 = (tile % ntN) << 4;

    int lane = threadIdx.x & 31;
    int half = lane >> 4;      // 0 or 1
    int l15  = lane & 15;

    // A fragment rows: lane m = l15 ; elems 0..7 -> K=half*8+i, 8..15 -> K=16+half*8+i
    const unsigned short* Arow = A + (size_t)(m0 + l15) * lda + (half << 3);
    // B fragment: lane n = l15 ; elems 0..7 -> K=half*8+i, 8..15 -> K=16+half*8+i  (W is row-major [N,Kd])
    const unsigned short* Wrow = W + (size_t)(n0 + l15) * ldw + (half << 3);

    v8f acc = {};
    for (int k0 = 0; k0 < Kd; k0 += 32) {
        FragAB a, bm;
        a.q[0]  = *(const uint4*)(Arow + k0);
        a.q[1]  = *(const uint4*)(Arow + k0 + 16);
        bm.q[0] = *(const uint4*)(Wrow + k0);
        bm.q[1] = *(const uint4*)(Wrow + k0 + 16);
        acc = __builtin_amdgcn_wmma_f32_16x16x32_bf16(
            false, a.v, false, bm.v, (short)0, acc, false, false);
    }

    int   gcol = n0 + l15;
    float bv   = bias ? bias[gcol] : 0.0f;
    #pragma unroll
    for (int j = 0; j < 8; ++j) {
        int   grow = m0 + (half << 3) + j;
        float v    = acc[j] + bv;
        if (relu)  v = fmaxf(v, 0.0f);
        if (resid) v += resid[(size_t)grow * N + gcol];
        size_t o = (size_t)grow * N + gcol;
        if (Cf)  Cf[o]  = v;
        if (Cbf) Cbf[o] = f2bf(v);
    }
}

// ---------------- fused K+V projection (shares A fragments) ----------------
// Ko[r,e] = sum_d A[r,d]*Wk[e,d] ; Vo[r,e] = sum_d A[r,d]*Wv[e,d]
// A is kvln [M,128] bf16; each wave owns one 16x16 tile of BOTH outputs, so
// the 67 MB/layer A tensor is streamed once instead of twice.
__global__ void __launch_bounds__(256)
kv_proj_wmma(const unsigned short* __restrict__ A,
             const unsigned short* __restrict__ Wk,
             const unsigned short* __restrict__ Wv,
             unsigned short* __restrict__ Ko,
             unsigned short* __restrict__ Vo, int M) {
    constexpr int ntN = HD_ >> 4;                      // 8 N-tiles
    int ntiles = (M >> 4) * ntN;
    int tile   = blockIdx.x * 8 + (threadIdx.x >> 5);
    if (tile >= ntiles) return;
    int m0 = (tile / ntN) << 4;
    int n0 = (tile % ntN) << 4;

    int lane = threadIdx.x & 31;
    int half = lane >> 4;
    int l15  = lane & 15;

    const unsigned short* Arow = A  + (size_t)(m0 + l15) * HD_ + (half << 3);
    const unsigned short* Wkr  = Wk + (size_t)(n0 + l15) * HD_ + (half << 3);
    const unsigned short* Wvr  = Wv + (size_t)(n0 + l15) * HD_ + (half << 3);

    v8f ak = {}, av = {};
    #pragma unroll
    for (int k0 = 0; k0 < HD_; k0 += 32) {
        FragAB a, bk, bv;
        a.q[0]  = *(const uint4*)(Arow + k0);
        a.q[1]  = *(const uint4*)(Arow + k0 + 16);
        bk.q[0] = *(const uint4*)(Wkr + k0);
        bk.q[1] = *(const uint4*)(Wkr + k0 + 16);
        bv.q[0] = *(const uint4*)(Wvr + k0);
        bv.q[1] = *(const uint4*)(Wvr + k0 + 16);
        ak = __builtin_amdgcn_wmma_f32_16x16x32_bf16(
            false, a.v, false, bk.v, (short)0, ak, false, false);
        av = __builtin_amdgcn_wmma_f32_16x16x32_bf16(
            false, a.v, false, bv.v, (short)0, av, false, false);
    }

    int gcol = n0 + l15;
    #pragma unroll
    for (int j = 0; j < 8; ++j) {
        size_t o = (size_t)(m0 + (half << 3) + j) * HD_ + gcol;
        Ko[o] = f2bf(ak[j]);
        Vo[o] = f2bf(av[j]);
    }
}

// ---------------- fused positional-encoding + LayerNorm over memories ------
// out[b,k,:] = LN(memories[b,k,layer,:] + PE[memory_indices[b,k],:]) as bf16
__global__ void __launch_bounds__(256)
kv_ln_kernel(const float* __restrict__ mem, const int* __restrict__ midx,
             const float* __restrict__ g, const float* __restrict__ bta,
             unsigned short* __restrict__ out, int layer) {
    int bk  = blockIdx.x;                       // b*K_ + k
    int tid = threadIdx.x;
    const float* row  = mem + ((size_t)bk * L_ + layer) * D_;
    float seqv = (float)(MAX_STEPS_ - 1 - midx[bk]);
    const float ln1e4 = 9.210340371976184f;     // ln(10000)

    float vals[8], s = 0.f, ss = 0.f;
    #pragma unroll
    for (int i = 0; i < 8; ++i) {
        int d = tid + i * 256;
        int j = (d < D_ / 2) ? d : d - D_ / 2;
        float inv = __expf(-(2.0f * (float)j / (float)D_) * ln1e4);
        float ang = seqv * inv;
        float pe  = (d < D_ / 2) ? __sinf(ang) : __cosf(ang);
        float v = row[d] + pe;
        vals[i] = v; s += v; ss += v * v;
    }
    __shared__ float rs[256], rq[256];
    rs[tid] = s; rq[tid] = ss; __syncthreads();
    for (int o = 128; o > 0; o >>= 1) {
        if (tid < o) { rs[tid] += rs[tid + o]; rq[tid] += rq[tid + o]; }
        __syncthreads();
    }
    float mean = rs[0] * (1.0f / D_);
    float var  = rq[0] * (1.0f / D_) - mean * mean;
    float rstd = rsqrtf(var + 1e-5f);
    unsigned short* orow = out + (size_t)bk * D_;
    #pragma unroll
    for (int i = 0; i < 8; ++i) {
        int d = tid + i * 256;
        orow[d] = f2bf((vals[i] - mean) * rstd * g[d] + bta[d]);
    }
}

// ---------------- LayerNorm over f32 rows -> bf16 ----------------
__global__ void __launch_bounds__(256)
ln_rows_kernel(const float* __restrict__ in, const float* __restrict__ g,
               const float* __restrict__ bta, unsigned short* __restrict__ out) {
    int r = blockIdx.x, tid = threadIdx.x;
    const float* row = in + (size_t)r * D_;
    float vals[8], s = 0.f, ss = 0.f;
    #pragma unroll
    for (int i = 0; i < 8; ++i) {
        int d = tid + i * 256;
        float v = row[d];
        vals[i] = v; s += v; ss += v * v;
    }
    __shared__ float rs[256], rq[256];
    rs[tid] = s; rq[tid] = ss; __syncthreads();
    for (int o = 128; o > 0; o >>= 1) {
        if (tid < o) { rs[tid] += rs[tid + o]; rq[tid] += rq[tid + o]; }
        __syncthreads();
    }
    float mean = rs[0] * (1.0f / D_);
    float var  = rq[0] * (1.0f / D_) - mean * mean;
    float rstd = rsqrtf(var + 1e-5f);
    unsigned short* orow = out + (size_t)r * D_;
    #pragma unroll
    for (int i = 0; i < 8; ++i) {
        int d = tid + i * 256;
        orow[d] = f2bf((vals[i] - mean) * rstd * g[d] + bta[d]);
    }
}

// ---------------- attention: one workgroup per (b, h) ----------------
// energy/softmax/weighted-sum are GEMVs (single query row) -> VALU + LDS
__global__ void __launch_bounds__(256)
attn_kernel(const float* __restrict__ q, const unsigned short* __restrict__ kp,
            const unsigned short* __restrict__ vp, const int* __restrict__ mask,
            unsigned short* __restrict__ o) {
    int bh = blockIdx.x;
    int b  = bh / H_, h = bh % H_;
    int tid = threadIdx.x, w = tid >> 5, lane = tid & 31;

    __shared__ float e[K_];
    __shared__ float red[8];
    __shared__ float osum[2][HD_];

    const float* qrow = q + (size_t)bh * HD_ + lane * 4;
    float q0 = qrow[0], q1 = qrow[1], q2 = qrow[2], q3 = qrow[3];

    for (int key = w; key < K_; key += 8) {
        const unsigned short* kr =
            kp + ((size_t)(b * K_ + key) * H_ + h) * HD_ + lane * 4;
        uint2 kk = *(const uint2*)kr;                 // 4 bf16
        float p = q0 * bf2f(kk.x) + q1 * bf2f(kk.x >> 16)
                + q2 * bf2f(kk.y) + q3 * bf2f(kk.y >> 16);
        #pragma unroll
        for (int off = 16; off; off >>= 1) p += __shfl_xor(p, off, 32);
        if (lane == 0) e[key] = p;
    }
    __syncthreads();

    const float sc = 0.022097086912079608f;           // 1/sqrt(2048)
    int   k0 = tid, k1 = tid + 256;
    float e0 = mask[b * K_ + k0] ? e[k0] * sc : -1e20f * sc;
    float e1 = mask[b * K_ + k1] ? e[k1] * sc : -1e20f * sc;

    // block max
    float m = fmaxf(e0, e1);
    #pragma unroll
    for (int off = 16; off; off >>= 1) m = fmaxf(m, __shfl_xor(m, off, 32));
    if (lane == 0) red[w] = m;
    __syncthreads();
    float mx = red[0];
    #pragma unroll
    for (int i = 1; i < 8; ++i) mx = fmaxf(mx, red[i]);
    __syncthreads();

    // block sum of exp
    float p0 = __expf(e0 - mx), p1 = __expf(e1 - mx);
    float s = p0 + p1;
    #pragma unroll
    for (int off = 16; off; off >>= 1) s += __shfl_xor(s, off, 32);
    if (lane == 0) red[w] = s;
    __syncthreads();
    float tot = 0.f;
    #pragma unroll
    for (int i = 0; i < 8; ++i) tot += red[i];
    float rtot = 1.0f / tot;
    e[k0] = p0 * rtot;
    e[k1] = p1 * rtot;
    __syncthreads();

    // o[b,h,d] = sum_k attn[k] * v[b,k,h,d]
    // all 256 threads: half-block `part` covers keys [part*256, part*256+256)
    {
        int part = tid >> 7;                 // 0 or 1
        int d    = tid & (HD_ - 1);
        float acc = 0.f;
        const unsigned short* vb =
            vp + ((size_t)(b * K_ + part * 256) * H_ + h) * HD_ + d;
        for (int k = 0; k < 256; ++k) {
            if ((k & 15) == 0)               // prefetch next chunk of V stream
                __builtin_prefetch(vb + (size_t)(k + 16) * H_ * HD_, 0, 0);
            acc += e[part * 256 + k] * bf2f(vb[(size_t)k * H_ * HD_]);
        }
        osum[part][d] = acc;
    }
    __syncthreads();
    if (tid < HD_)
        o[(size_t)bh * HD_ + tid] = f2bf(osum[0][tid] + osum[1][tid]);
}

// ---------------- actor/critic heads ----------------
__global__ void __launch_bounds__(256)
heads_kernel(const float* __restrict__ hid, const float* __restrict__ aw,
             const float* __restrict__ ab, const float* __restrict__ cw,
             const float* __restrict__ cb, float* __restrict__ out) {
    int b = blockIdx.x, tid = threadIdx.x;
    __shared__ float red[256];
    const float* hr = hid + (size_t)b * D_;
    for (int j = 0; j <= A_; ++j) {
        const float* wr = (j < A_) ? aw + (size_t)j * D_ : cw;
        float p = 0.f;
        for (int d = tid; d < D_; d += 256) p += hr[d] * wr[d];
        red[tid] = p; __syncthreads();
        for (int o = 128; o > 0; o >>= 1) {
            if (tid < o) red[tid] += red[tid + o];
            __syncthreads();
        }
        if (tid == 0)
            out[b * (A_ + 1) + j] = red[0] + ((j < A_) ? ab[j] : cb[0]);
        __syncthreads();
    }
}

// ==================== host-side orchestration ====================
extern "C" void kernel_launch(void* const* d_in, const int* in_sizes, int n_in,
                              void* d_out, int out_size, void* d_ws, size_t ws_size,
                              hipStream_t stream) {
    const float* x        = (const float*)d_in[0];
    const float* memories = (const float*)d_in[1];
    const int*   mask     = (const int*)d_in[2];
    const int*   midx     = (const int*)d_in[3];
    const float* enc_w    = (const float*)d_in[4];
    const float* enc_b    = (const float*)d_in[5];
    const float* lnq_g    = (const float*)d_in[6];
    const float* lnq_b    = (const float*)d_in[7];
    const float* lnkv_g   = (const float*)d_in[8];
    const float* lnkv_b   = (const float*)d_in[9];
    const float* lnat_g   = (const float*)d_in[10];
    const float* lnat_b   = (const float*)d_in[11];
    const float* wq       = (const float*)d_in[12];
    const float* wk       = (const float*)d_in[13];
    const float* wv       = (const float*)d_in[14];
    const float* wo       = (const float*)d_in[15];
    const float* bo       = (const float*)d_in[16];
    const float* fc_w     = (const float*)d_in[17];
    const float* fc_b     = (const float*)d_in[18];
    const float* post_w   = (const float*)d_in[19];
    const float* post_b   = (const float*)d_in[20];
    const float* actor_w  = (const float*)d_in[21];
    const float* actor_b  = (const float*)d_in[22];
    const float* critic_w = (const float*)d_in[23];
    const float* critic_b = (const float*)d_in[24];
    float* out = (float*)d_out;

    // -------- workspace carve-out --------
    char*  base = (char*)d_ws;
    size_t off  = 0;
    auto alloc = [&](size_t bytes) -> void* {
        void* p = base + off;
        off = (off + bytes + 255) & ~(size_t)255;
        return p;
    };
    unsigned short* wq16   = (unsigned short*)alloc((size_t)L_ * HD_ * HD_ * 2);
    unsigned short* wk16   = (unsigned short*)alloc((size_t)L_ * HD_ * HD_ * 2);
    unsigned short* wv16   = (unsigned short*)alloc((size_t)L_ * HD_ * HD_ * 2);
    unsigned short* wo16   = (unsigned short*)alloc((size_t)L_ * D_ * D_ * 2);
    unsigned short* fcw16  = (unsigned short*)alloc((size_t)L_ * D_ * D_ * 2);
    unsigned short* pw16   = (unsigned short*)alloc((size_t)D_ * D_ * 2);
    float*          h32    = (float*)alloc((size_t)B_ * D_ * 4);
    unsigned short* qln16  = (unsigned short*)alloc((size_t)B_ * D_ * 2);
    float*          q32    = (float*)alloc((size_t)B_ * D_ * 4);
    unsigned short* kvln16 = (unsigned short*)alloc((size_t)B_ * K_ * D_ * 2);
    unsigned short* kp16   = (unsigned short*)alloc((size_t)B_ * K_ * D_ * 2);
    unsigned short* vp16   = (unsigned short*)alloc((size_t)B_ * K_ * D_ * 2);
    unsigned short* o16    = (unsigned short*)alloc((size_t)B_ * D_ * 2);
    float*          xr32   = (float*)alloc((size_t)B_ * D_ * 4);
    unsigned short* xrl16  = (unsigned short*)alloc((size_t)B_ * D_ * 2);
    unsigned short* h16    = (unsigned short*)alloc((size_t)B_ * D_ * 2);
    float*          hid32  = (float*)alloc((size_t)B_ * D_ * 4);
    (void)ws_size; (void)n_in; (void)in_sizes; (void)out_size;

    auto cvt = [&](const float* src, unsigned short* dst, size_t n) {
        cvt_bf16_kernel<<<dim3((unsigned)((n + 255) / 256)), dim3(256), 0, stream>>>(src, dst, n);
    };
    // weights -> bf16 (must be redone each call; no persistent state allowed)
    cvt(wq, wq16, (size_t)L_ * HD_ * HD_);
    cvt(wk, wk16, (size_t)L_ * HD_ * HD_);
    cvt(wv, wv16, (size_t)L_ * HD_ * HD_);
    cvt(wo, wo16, (size_t)L_ * D_ * D_);
    cvt(fc_w, fcw16, (size_t)L_ * D_ * D_);
    cvt(post_w, pw16, (size_t)D_ * D_);

    // encoder
    encoder_kernel<<<dim3((B_ * D_ + 255) / 256), dim3(256), 0, stream>>>(x, enc_w, enc_b, h32);

    auto gemm = [&](const unsigned short* A, int lda, const unsigned short* W, int ldw,
                    float* Cf, const float* bias, const float* resid,
                    unsigned short* Cbf, int M, int N, int Kd, int relu) {
        int ntiles = (M / 16) * (N / 16);
        gemm_bf16_wmma<<<dim3((ntiles + 7) / 8), dim3(256), 0, stream>>>(
            A, lda, W, ldw, Cf, bias, resid, Cbf, M, N, Kd, relu);
    };

    for (int l = 0; l < L_; ++l) {
        // q path: LN(h) then per-head projection (shared Wq across heads/batch)
        ln_rows_kernel<<<dim3(B_), dim3(256), 0, stream>>>(
            h32, lnq_g + (size_t)l * D_, lnq_b + (size_t)l * D_, qln16);
        gemm(qln16, HD_, wq16 + (size_t)l * HD_ * HD_, HD_,
             q32, nullptr, nullptr, nullptr, B_ * H_, HD_, HD_, 0);

        // kv path: PE + LN fused, then fused K+V projection (single A stream)
        kv_ln_kernel<<<dim3(B_ * K_), dim3(256), 0, stream>>>(
            memories, midx, lnkv_g + (size_t)l * D_, lnkv_b + (size_t)l * D_, kvln16, l);
        {
            int M = B_ * K_ * H_;
            int ntiles = (M / 16) * (HD_ / 16);
            kv_proj_wmma<<<dim3((ntiles + 7) / 8), dim3(256), 0, stream>>>(
                kvln16, wk16 + (size_t)l * HD_ * HD_, wv16 + (size_t)l * HD_ * HD_,
                kp16, vp16, M);
        }

        // attention (per (b,h) GEMV + softmax)
        attn_kernel<<<dim3(B_ * H_), dim3(256), 0, stream>>>(q32, kp16, vp16, mask, o16);

        // fc_out + residual
        gemm(o16, D_, wo16 + (size_t)l * D_ * D_, D_,
             xr32, bo + (size_t)l * D_, h32, nullptr, B_, D_, D_, 0);

        // LN + MLP (relu) + residual -> new h (f32 + bf16 copies)
        ln_rows_kernel<<<dim3(B_), dim3(256), 0, stream>>>(
            xr32, lnat_g + (size_t)l * D_, lnat_b + (size_t)l * D_, xrl16);
        gemm(xrl16, D_, fcw16 + (size_t)l * D_ * D_, D_,
             h32, fc_b + (size_t)l * D_, xr32, h16, B_, D_, D_, 1);
    }

    // post head: hid = relu(h @ post_w.T + post_b)
    gemm(h16, D_, pw16, D_, hid32, post_b, nullptr, nullptr, B_, D_, D_, 1);

    // actor + critic heads -> [B, A+1]
    heads_kernel<<<dim3(B_), dim3(256), 0, stream>>>(
        hid32, actor_w, actor_b, critic_w, critic_b, out);
}
